// Hebb_VGG_Channel_Boost_3513283248181
// MI455X (gfx1250) — compile-verified, matrix operand order checked by reading, not run on hardware
//
#include <hip/hip_runtime.h>
#include <hip/hip_bf16.h>

typedef __attribute__((ext_vector_type(16))) _Float16 v16h;
typedef __attribute__((ext_vector_type(8)))  float    v8f;
typedef __attribute__((ext_vector_type(4)))  float    v4f;

#define C_CH      512
#define HW        200704           // 448*448
#define HW4       50176            // HW/4 (float4 units per channel)
#define BLK_PER_C 196              // HW4 / 256
#define OUT_ELEMS 102760448ULL     // 512*448*448

// ---------------------------------------------------------------------------
// Kernel 1: per-channel mean over H*W.  One block per channel, 1024 threads,
// each thread does exactly 49 float4 loads (49*1024*4 == 200704).  Pure
// bandwidth pass, LDS tree reduction, no atomics (deterministic).
// ---------------------------------------------------------------------------
__global__ void __launch_bounds__(1024)
channel_mean_kernel(const float4* __restrict__ x4, float* __restrict__ avr) {
    __shared__ float red[1024];
    const int c = blockIdx.x;
    const int t = threadIdx.x;
    const size_t base = (size_t)c * HW4;
    float s = 0.0f;
#pragma unroll 7
    for (int i = 0; i < 49; ++i) {
        float4 v = x4[base + (size_t)i * 1024 + t];
        s += (v.x + v.y) + (v.z + v.w);
    }
    red[t] = s;
    __syncthreads();
    for (int off = 512; off > 0; off >>= 1) {
        if (t < off) red[t] += red[t + off];
        __syncthreads();
    }
    if (t == 0) avr[c] = red[0] * (1.0f / (float)HW);
}

// ---------------------------------------------------------------------------
// Kernel 2: the whole "head" in one 512-thread (16-wave) workgroup.
//   boost_vec = W @ x_avr        -> per-wave WMMA f32_16x16x32_f16
//   inh/relu/threshold/sparse    -> LDS tree reductions
//   Frobenius norm closed-form:  ||outer(a,b) \ diag||^2 = (Sa2)(Sb2) - S(a2b2)
//   new_boost = 0.5*W + 0.05*outer(x_tmp, x_sparse)/norm  (coalesced write)
//   scale[c]  = x_tmp[c] / (x_avr[c] + 1e-12)             -> workspace
//
// 16-bit A-matrix 16x32 layout (ISA 7.12.2): lane L holds row M = L%16; the
// 16 K-values per lane are two contiguous 8-runs:
//   K in [k0+off, k0+off+8)   -> halves a[0..7]
//   K in [k0+off+16, ..+8)    -> halves a[8..15]
// with off = 8 for lanes 16..31.  B (32x16) mirrors this with lane -> column.
// ---------------------------------------------------------------------------
__device__ __forceinline__ float blk_sum(float v, float* s, int t) {
    s[t] = v;
    __syncthreads();
    for (int off = 256; off > 0; off >>= 1) {
        if (t < off) s[t] += s[t + off];
        __syncthreads();
    }
    float r = s[0];
    __syncthreads();
    return r;
}

__device__ __forceinline__ float blk_max(float v, float* s, int t) {
    s[t] = v;
    __syncthreads();
    for (int off = 256; off > 0; off >>= 1) {
        if (t < off) s[t] = fmaxf(s[t], s[t + off]);
        __syncthreads();
    }
    float r = s[0];
    __syncthreads();
    return r;
}

__global__ void __launch_bounds__(512)
head_kernel(const float* __restrict__ W,      // boost_weight [512,512]
            float* __restrict__ ws,           // [0..511]=x_avr in, [512..1023]=scale out
            float* __restrict__ nb_out) {     // new_boost_weight [512,512]
    __shared__ float s_avr[C_CH];
    __shared__ float s_bv[C_CH];
    __shared__ float s_xt[C_CH];
    __shared__ float s_red[C_CH];

    const int t = threadIdx.x;
    s_avr[t] = ws[t];
    __syncthreads();

    // ---- boost_vec = W @ x_avr via per-wave WMMA ----
    const int   wv   = t >> 5;              // wave 0..15 -> rows 16wv..16wv+15
    const int   lane = t & 31;
    const int   mrow = (wv << 4) + (lane & 15);
    const int   koff = (lane & 16) ? 8 : 0; // lane-half K offset
    const float bm   = ((lane & 15) == 0) ? 1.0f : 0.0f;  // B col N==0 mask

    const float* wrow = W + (size_t)mrow * C_CH;

    v8f acc = {0.f, 0.f, 0.f, 0.f, 0.f, 0.f, 0.f, 0.f};
    for (int kk = 0; kk < 16; ++kk) {       // 16 K-tiles of 32 cover K=512
        const int b1 = (kk << 5) + koff;    // first contiguous 8-run
        const int b2 = b1 + 16;             // second contiguous 8-run

        // A: 4x global b128 (row-contiguous), convert f32 -> f16
        const float4 wa0 = *(const float4*)(wrow + b1);
        const float4 wa1 = *(const float4*)(wrow + b1 + 4);
        const float4 wa2 = *(const float4*)(wrow + b2);
        const float4 wa3 = *(const float4*)(wrow + b2 + 4);
        // B: 4x ds b128, masked so only column N==0 is nonzero
        const float4 vb0 = *(const float4*)(s_avr + b1);
        const float4 vb1 = *(const float4*)(s_avr + b1 + 4);
        const float4 vb2 = *(const float4*)(s_avr + b2);
        const float4 vb3 = *(const float4*)(s_avr + b2 + 4);

        v16h a, b;
        a[0]=(_Float16)wa0.x; a[1]=(_Float16)wa0.y; a[2]=(_Float16)wa0.z; a[3]=(_Float16)wa0.w;
        a[4]=(_Float16)wa1.x; a[5]=(_Float16)wa1.y; a[6]=(_Float16)wa1.z; a[7]=(_Float16)wa1.w;
        a[8]=(_Float16)wa2.x; a[9]=(_Float16)wa2.y; a[10]=(_Float16)wa2.z; a[11]=(_Float16)wa2.w;
        a[12]=(_Float16)wa3.x; a[13]=(_Float16)wa3.y; a[14]=(_Float16)wa3.z; a[15]=(_Float16)wa3.w;

        b[0]=(_Float16)(bm*vb0.x); b[1]=(_Float16)(bm*vb0.y); b[2]=(_Float16)(bm*vb0.z); b[3]=(_Float16)(bm*vb0.w);
        b[4]=(_Float16)(bm*vb1.x); b[5]=(_Float16)(bm*vb1.y); b[6]=(_Float16)(bm*vb1.z); b[7]=(_Float16)(bm*vb1.w);
        b[8]=(_Float16)(bm*vb2.x); b[9]=(_Float16)(bm*vb2.y); b[10]=(_Float16)(bm*vb2.z); b[11]=(_Float16)(bm*vb2.w);
        b[12]=(_Float16)(bm*vb3.x); b[13]=(_Float16)(bm*vb3.y); b[14]=(_Float16)(bm*vb3.z); b[15]=(_Float16)(bm*vb3.w);

        acc = __builtin_amdgcn_wmma_f32_16x16x32_f16(
                  false, a, false, b, (short)0, acc, false, false);
    }
    // C/D layout: VGPR j -> (M=j, N=lane) lanes 0-15, (M=j+8, N=lane-16)
    // lanes 16-31.  Column N==0 lives in lanes 0 and 16.
    if ((lane & 15) == 0) {
        const int mbase = (wv << 4) + ((lane & 16) ? 8 : 0);
#pragma unroll
        for (int j = 0; j < 8; ++j) s_bv[mbase + j] = acc[j];
    }
    __syncthreads();

    // ---- inhibition / relu / threshold / sparsify ----
    const float avr = s_avr[t];
    const float bv  = s_bv[t];
    const float inh = blk_sum(bv, s_red, t) * (1.0f / (float)C_CH);
    float xt = avr + bv - 2.0f * inh;          // INH_C = 2.0
    xt = xt > 0.0f ? xt : 0.0f;                // relu
    const float thr = blk_max(xt, s_red, t) * 0.9f;   // SPARSE_THRES
    const float xs  = (xt < thr) ? 0.0f : xt;

    // ---- Frobenius norm of zero-diagonal outer product, closed form ----
    const float a2 = blk_sum(xt * xt, s_red, t);
    const float b2 = blk_sum(xs * xs, s_red, t);
    const float c2 = blk_sum(xt * xt * xs * xs, s_red, t);
    const float nrm = sqrtf(a2 * b2 - c2);
    const float inv = (nrm > 0.0f && !__builtin_isnan(nrm) && !__builtin_isinf(nrm))
                          ? (1.0f / nrm) : 1.0f;

    s_xt[t] = xt;
    ws[C_CH + t] = xt / (avr + 1e-12f);        // per-channel scale for pass 3
    __syncthreads();

    // ---- new_boost = 0.5*W + 0.05*outer(xt, xs)/norm  (diag zeroed) ----
    // Thread t owns column t, loop over rows -> fully coalesced 2KB stores.
    for (int r = 0; r < C_CH; ++r) {
        const float wt = (r == t) ? 0.0f : s_xt[r] * xs * inv;
        nb_out[(size_t)r * C_CH + t] = W[(size_t)r * C_CH + t] * 0.5f + 0.05f * wt;
    }
}

// ---------------------------------------------------------------------------
// Kernel 3: out = x * scale[c].  One float4 per thread; 196 blocks per
// channel exactly, so the channel index (and scale load) is block-uniform.
// Non-temporal store for `out` (never re-read) to keep L2 for x.
// ---------------------------------------------------------------------------
__global__ void __launch_bounds__(256)
scale_kernel(const float4* __restrict__ x4,
             const float* __restrict__ scale,
             float* __restrict__ out) {
    const int b = blockIdx.x;
    const int c = b / BLK_PER_C;               // uniform per block -> scalar
    const float sc = scale[c];
    const size_t i = (size_t)b * 256 + threadIdx.x;
    float4 v = x4[i];
    v4f r = {v.x * sc, v.y * sc, v.z * sc, v.w * sc};
    __builtin_nontemporal_store(r, (v4f*)(out + i * 4));
}

// ---------------------------------------------------------------------------
extern "C" void kernel_launch(void* const* d_in, const int* in_sizes, int n_in,
                              void* d_out, int out_size, void* d_ws, size_t ws_size,
                              hipStream_t stream) {
    const float* x = (const float*)d_in[0];       // [1,512,448,448]
    const float* W = (const float*)d_in[1];       // [512,512]
    float* out = (float*)d_out;                   // out (411MB) ++ new_boost (1MB)
    float* ws  = (float*)d_ws;                    // [0..511]=avr, [512..1023]=scale

    channel_mean_kernel<<<C_CH, 1024, 0, stream>>>((const float4*)x, ws);
    head_kernel<<<1, 512, 0, stream>>>(W, ws, out + OUT_ELEMS);
    scale_kernel<<<C_CH * BLK_PER_C, 256, 0, stream>>>((const float4*)x, ws + C_CH, out);
}
